// MAM_68831145885814
// MI455X (gfx1250) — compile-verified, matrix operand order checked by reading, not run on hardware
//
#include <hip/hip_runtime.h>
#include <hip/hip_bf16.h>
#include <math.h>

// ---------------------------------------------------------------------------
// MI455X / gfx1250 implementation.
// All convolutions = implicit-GEMM with V_WMMA_F32_16X16X32_BF16 (f32 accum),
// BN+bias+ReLU fused in the epilogue. The 85MB per-pixel dynamic-filter tensor
// is kept entirely in LDS (fused GEMM + kernel2d apply). Double-buffered B
// staging overlaps global gather latency with the WMMA pipe.
// ---------------------------------------------------------------------------

typedef __bf16 bf16;
typedef __attribute__((ext_vector_type(16))) __bf16 v16bf;
typedef __attribute__((ext_vector_type(8)))  float  v8f;

#define BN_EPS 1e-5f

// ======================= small elementwise kernels ==========================

__global__ void k_cvt(const float* __restrict__ s, bf16* __restrict__ d, long n) {
  long i = (long)blockIdx.x * blockDim.x + threadIdx.x;
  if (i < n) d[i] = (bf16)s[i];
}

// BN folding: sc = g*rsqrt(v+eps); bs = (b - m)*sc + beta
__global__ void k_bncoef(const float* __restrict__ b, const float* __restrict__ g,
                         const float* __restrict__ be, const float* __restrict__ m,
                         const float* __restrict__ v, float* __restrict__ sc,
                         float* __restrict__ bs, int C) {
  int c = blockIdx.x * blockDim.x + threadIdx.x;
  if (c < C) {
    float s = g[c] * rsqrtf(v[c] + BN_EPS);
    sc[c] = s;
    bs[c] = (b[c] - m[c]) * s + be[c];
  }
}

// gr/gt f32 -> bf16 copies, plus x0=concat(gr,gt) and cat1[ch0:128]=gr
__global__ void k_prep(const float* __restrict__ gr, const float* __restrict__ gt,
                       bf16* __restrict__ gr_b, bf16* __restrict__ gt_b,
                       bf16* __restrict__ x0, bf16* __restrict__ cat1,
                       long per, long chw) {
  long i = (long)blockIdx.x * blockDim.x + threadIdx.x;
  if (i >= per) return;
  long n = i / chw, rem = i % chw;
  bf16 a = (bf16)gr[i], b = (bf16)gt[i];
  gr_b[i] = a; gt_b[i] = b;
  long x0g = n * 2 * chw + rem;
  x0[x0g] = a; x0[x0g + chw] = b;
  cat1[x0g] = a;                       // cat1 second half filled by k_warp
}

// ======================= generic WMMA conv (CBR) ============================
// Y[oc,pix] = relu( sc[oc] * (W @ im2col(X))[oc,pix] + bs[oc] )
// One wave -> one 16(oc) x 16(pixel) tile. Software-pipelined, double-buffered
// B staging: gather chunk i+1 into regs, WMMA chunk i, then spill regs to LDS.
template <int KS, int STRIDE, int PAD, bool RELU>
__global__ void k_conv_wmma(const bf16* __restrict__ in, bf16* __restrict__ outb,
                            float* __restrict__ outf, const bf16* __restrict__ W,
                            const float* __restrict__ sc, const float* __restrict__ bs,
                            int Nn, int IC, int OC, int IH, int IW, int OH, int OW,
                            int outCtot, int ocOff) {
  const int K    = IC * KS * KS;                 // multiple of 32 for all layers
  const int lane = threadIdx.x & 31;
  const int wv   = threadIdx.x >> 5;
  const int nWav = blockDim.x >> 5;
  const int pixBase = blockIdx.x * 16;
  const int ocBase  = (blockIdx.y * nWav + wv) * 16;
  const int NPIX = Nn * OH * OW;

  __shared__ bf16 ldsB[2][16][34];               // double-buffered [pix][k]
  __shared__ int  pN[16], pY[16], pX[16];        // pixel decomposition (once)

  if (threadIdx.x < 16) {
    int pix = pixBase + threadIdx.x;
    if (pix >= NPIX) pix = NPIX - 1;
    int n = pix / (OH * OW), r = pix % (OH * OW);
    pN[threadIdx.x] = n; pY[threadIdx.x] = r / OW; pX[threadIdx.x] = r % OW;
  }
  __syncthreads();

  bf16 regs[4];                                   // 512 elems / >=128 threads
  auto gatherB = [&](int k0) {
    int c = 0;
    for (int e = threadIdx.x; e < 512; e += blockDim.x, ++c) {
      int p = e >> 5, kk = e & 31, k = k0 + kk;
      bf16 val = (bf16)0.f;
      if (k < K) {
        int ic = k / (KS * KS), rr = k % (KS * KS);
        int iy = pY[p] * STRIDE - PAD + rr / KS;
        int ix = pX[p] * STRIDE - PAD + rr % KS;
        if (iy >= 0 && iy < IH && ix >= 0 && ix < IW)
          val = in[(((long)pN[p] * IC + ic) * IH + iy) * IW + ix];
      }
      regs[c] = val;
    }
  };
  auto storeB = [&](int buf) {
    int c = 0;
    for (int e = threadIdx.x; e < 512; e += blockDim.x, ++c)
      ldsB[buf][e >> 5][e & 31] = regs[c];
  };

  gatherB(0); storeB(0);

  const int kb = (lane >> 4) * 8;                 // lane-half K split (ISA layout)
  const int mn = lane & 15;
  const int nch = K / 32;
  v8f acc = {};
  for (int i = 0; i < nch; ++i) {
    if (i + 1 < nch) gatherB((i + 1) * 32);       // issue global loads
    __syncthreads();                              // buf i&1 ready for all waves
    const bf16* wrow = W + (long)(ocBase + mn) * K + i * 32;
    __builtin_prefetch(wrow + 64, 0, 0);          // global_prefetch_b8
    v16bf a, b;
#pragma unroll
    for (int r = 0; r < 8; ++r) {
      int kk = ((r < 4) ? 2 * r : 2 * r + 8) + kb;
      a[2 * r]     = wrow[kk];
      a[2 * r + 1] = wrow[kk + 1];
      b[2 * r]     = ldsB[i & 1][mn][kk];
      b[2 * r + 1] = ldsB[i & 1][mn][kk + 1];
    }
    acc = __builtin_amdgcn_wmma_f32_16x16x32_bf16(false, a, false, b,
                                                  (short)0, acc, false, false);
    if (i + 1 < nch) storeB((i + 1) & 1);         // s_wait_loadcnt lands here
  }

  // epilogue: C/D layout -> (oc, pixel), fused BN + ReLU
  const int pix = pixBase + (lane & 15);
  const int mo  = (lane >> 4) * 8;
  if (pix < NPIX) {
    int n = pix / (OH * OW), r = pix % (OH * OW);
    int oh = r / OW, ow = r % OW;
#pragma unroll
    for (int rr = 0; rr < 8; ++rr) {
      int oc = ocBase + mo + rr;
      if (oc < OC) {
        float y = acc[rr] * sc[oc] + bs[oc];
        if (RELU) y = y > 0.f ? y : 0.f;
        long o = (((long)n * outCtot + ocOff + oc) * OH + oh) * OW + ow;
        if (outb) outb[o] = (bf16)y; else outf[o] = y;
      }
    }
  }
}

// ============ fused dynamic-filter GEMM + per-pixel 3x3 conv ================
// Block = 16 pixels, 8 waves, each wave does 9 oc-tiles -> all 1152 filter
// channels live in LDS; then out[c,p] = sum_j gt_patch[c,j,p]*F[c*9+j,p] + gr.
// All 9 A-fragments are loaded as one clause BEFORE the 9 WMMAs so the matrix
// pipe overlaps the weight loads; B staging is double-buffered like above.
__global__ void k_dyn_fused(const bf16* __restrict__ gr, const bf16* __restrict__ gt,
                            const bf16* __restrict__ Wd /* [1152][1152] */,
                            const float* __restrict__ bd, bf16* __restrict__ dynout,
                            int Nn, int Hh, int Ww) {
  const int IC = 128, OC = 1152, K = 1152;
  extern __shared__ char smem[];
  float* ldsF = (float*)smem;                            // [1152][16] f32
  bf16*  ldsB = (bf16*)(smem + OC * 16 * sizeof(float)); // [2][16][34]
  __shared__ int pN[16], pY[16], pX[16];

  const int lane = threadIdx.x & 31;
  const int wv   = threadIdx.x >> 5;
  const int pixBase = blockIdx.x * 16;
  const int NPIX = Nn * Hh * Ww;

  if (threadIdx.x < 16) {
    int pix = pixBase + threadIdx.x;
    if (pix >= NPIX) pix = NPIX - 1;
    int n = pix / (Hh * Ww), r = pix % (Hh * Ww);
    pN[threadIdx.x] = n; pY[threadIdx.x] = r / Ww; pX[threadIdx.x] = r % Ww;
  }
  __syncthreads();

  bf16 regs[2];
  auto gatherB = [&](int k0) {
    int c = 0;
    for (int e = threadIdx.x; e < 512; e += blockDim.x, ++c) {
      int p = e >> 5, kk = e & 31, k = k0 + kk;
      bf16 val = (bf16)0.f;
      int ic = k / 9, rr = k % 9;
      int iy = pY[p] - 1 + rr / 3, ix = pX[p] - 1 + rr % 3;
      if (iy >= 0 && iy < Hh && ix >= 0 && ix < Ww)
        val = gr[(((long)pN[p] * IC + ic) * Hh + iy) * Ww + ix];
      regs[c] = val;
    }
  };
  auto storeB = [&](int buf) {
    int c = 0;
    for (int e = threadIdx.x; e < 512; e += blockDim.x, ++c)
      ldsB[(long)buf * 16 * 34 + (e >> 5) * 34 + (e & 31)] = regs[c];
  };

  gatherB(0); storeB(0);

  const int kb = (lane >> 4) * 8;
  const int mn = lane & 15;
  v8f acc[9];
#pragma unroll
  for (int t = 0; t < 9; ++t) acc[t] = (v8f){};

  const int nch = K / 32;
  for (int i = 0; i < nch; ++i) {
    if (i + 1 < nch) gatherB((i + 1) * 32);
    __syncthreads();
    // B fragment once per chunk
    v16bf b;
#pragma unroll
    for (int r = 0; r < 8; ++r) {
      int kk = ((r < 4) ? 2 * r : 2 * r + 8) + kb;
      b[2 * r]     = ldsB[(i & 1) * 16 * 34 + mn * 34 + kk];
      b[2 * r + 1] = ldsB[(i & 1) * 16 * 34 + mn * 34 + kk + 1];
    }
    // hoist ALL 9 A-fragment loads (one load clause), then 9 back-to-back WMMAs
    v16bf a[9];
#pragma unroll
    for (int t = 0; t < 9; ++t) {
      const bf16* wrow = Wd + (long)((wv * 9 + t) * 16 + mn) * K + i * 32;
#pragma unroll
      for (int r = 0; r < 8; ++r) {
        int kk = ((r < 4) ? 2 * r : 2 * r + 8) + kb;
        a[t][2 * r]     = wrow[kk];
        a[t][2 * r + 1] = wrow[kk + 1];
      }
    }
#pragma unroll
    for (int t = 0; t < 9; ++t)
      acc[t] = __builtin_amdgcn_wmma_f32_16x16x32_bf16(false, a[t], false, b,
                                                       (short)0, acc[t], false, false);
    if (i + 1 < nch) storeB((i + 1) & 1);
  }

  // spill filter tile (with dyn bias) to LDS
  {
    const int p  = lane & 15;
    const int mo = (lane >> 4) * 8;
#pragma unroll
    for (int t = 0; t < 9; ++t) {
      const int ocBase = (wv * 9 + t) * 16;
#pragma unroll
      for (int rr = 0; rr < 8; ++rr) {
        int oc = ocBase + mo + rr;
        ldsF[oc * 16 + p] = acc[t][rr] + bd[oc];
      }
    }
  }
  __syncthreads();

  // apply per-pixel 3x3 filter to gt (edge-replicate), add gr residual
  for (int o = threadIdx.x; o < 16 * IC; o += blockDim.x) {
    int c = o >> 4, p = o & 15;
    int pix = pixBase + p;
    if (pix >= NPIX) continue;
    int oh = pY[p], ow = pX[p];
    long cb = ((long)pN[p] * IC + c) * Hh;
    float s = (float)gr[(cb + oh) * Ww + ow];
#pragma unroll
    for (int j = 0; j < 9; ++j) {
      int iy = oh - 1 + j / 3, ix = ow - 1 + j % 3;
      iy = iy < 0 ? 0 : (iy >= Hh ? Hh - 1 : iy);   // replicate pad
      ix = ix < 0 ? 0 : (ix >= Ww ? Ww - 1 : ix);
      s += (float)gt[(cb + iy) * Ww + ix] * ldsF[(c * 9 + j) * 16 + p];
    }
    dynout[(cb + oh) * Ww + ow] = (bf16)s;
  }
}

// =========================== small stage kernels ============================

__global__ void k_pool(const bf16* __restrict__ n3, float* __restrict__ pooled,
                       int Nn, int C, int S) {
  int i = blockIdx.x * blockDim.x + threadIdx.x;
  if (i >= Nn * C) return;
  const bf16* p = n3 + (long)i * S * S;
  float s = 0.f;
  for (int k = 0; k < S * S; ++k) s += (float)p[k];
  pooled[i] = s / (float)(S * S);
}

__global__ void k_theta(const float* __restrict__ pooled, const float* __restrict__ w,
                        const float* __restrict__ b, float* __restrict__ theta, int Nn) {
  int t = threadIdx.x;
  if (t >= Nn * 6) return;
  int n = t / 6, j = t % 6;
  float s = b[j];
  for (int c = 0; c < 128; ++c) s += pooled[n * 128 + c] * w[j * 128 + c];
  const float iden[6] = {1.f, 0.f, 0.f, 0.f, 1.f, 0.f};
  theta[t] = s + iden[j];
}

// STN bilinear warp (zeros padding, align_corners=True); writes cat1 ch 128..255
__global__ void k_warp(const float* __restrict__ gt, const float* __restrict__ theta,
                       bf16* __restrict__ cat1, int Nn, int C, int Hh, int Ww) {
  int i = blockIdx.x * blockDim.x + threadIdx.x;
  if (i >= Nn * Hh * Ww) return;
  int n = i / (Hh * Ww), r = i % (Hh * Ww);
  int h = r / Ww, w = r % Ww;
  const float* th = theta + n * 6;
  float xx = -1.f + 2.f * w / (float)(Ww - 1);
  float yy = -1.f + 2.f * h / (float)(Hh - 1);
  float gx = th[0] * xx + th[1] * yy + th[2];
  float gy = th[3] * xx + th[4] * yy + th[5];
  float ix = (gx + 1.f) * 0.5f * (Ww - 1);
  float iy = (gy + 1.f) * 0.5f * (Hh - 1);
  int x0 = (int)floorf(ix), y0 = (int)floorf(iy);
  float wx1 = ix - x0, wy1 = iy - y0, wx0 = 1.f - wx1, wy0 = 1.f - wy1;
  int   xs[2] = {x0, x0 + 1}, ys[2] = {y0, y0 + 1};
  float wxt[2] = {wx0, wx1},  wyt[2] = {wy0, wy1};
  float wgtc[4]; int xi[4], yi[4];
  for (int q = 0; q < 4; ++q) {
    int xq = xs[q & 1], yq = ys[q >> 1];
    bool ok = xq >= 0 && xq < Ww && yq >= 0 && yq < Hh;
    wgtc[q] = ok ? wxt[q & 1] * wyt[q >> 1] : 0.f;
    xi[q] = xq < 0 ? 0 : (xq >= Ww ? Ww - 1 : xq);
    yi[q] = yq < 0 ? 0 : (yq >= Hh ? Hh - 1 : yq);
  }
  for (int c = 0; c < C; ++c) {
    long cb = ((long)n * C + c) * Hh;
    float v = 0.f;
    for (int q = 0; q < 4; ++q) v += gt[(cb + yi[q]) * Ww + xi[q]] * wgtc[q];
    cat1[(((long)n * 2 * C + C + c) * Hh + h) * Ww + w] = (bf16)v;
  }
}

// mod = gr + alpha*gt + bata with alpha/bata = per-pixel 1x1 convs (1 channel)
__global__ void k_mod(const float* __restrict__ gr, const float* __restrict__ gt,
                      const float* __restrict__ wa, const float* __restrict__ ba,
                      const float* __restrict__ wb, const float* __restrict__ bb,
                      bf16* __restrict__ mod, int Nn, int C, int Hh, int Ww) {
  int i = blockIdx.x * blockDim.x + threadIdx.x;
  if (i >= Nn * Hh * Ww) return;
  int n = i / (Hh * Ww), r = i % (Hh * Ww);
  long base = (long)n * C * Hh * Ww + r;
  long hw = (long)Hh * Ww;
  float a = ba[0], bsc = bb[0];
  for (int c = 0; c < C; ++c) {
    float g = gr[base + c * hw];
    a += wa[c] * g; bsc += wb[c] * g;
  }
  for (int c = 0; c < C; ++c)
    mod[base + c * hw] = (bf16)(gr[base + c * hw] + a * gt[base + c * hw] + bsc);
}

// ================================ host side =================================

static inline char* carve(char*& p, size_t bytes) {
  char* r = p; p += (bytes + 255) & ~(size_t)255; return r;
}

extern "C" void kernel_launch(void* const* d_in, const int* in_sizes, int n_in,
                              void* d_out, int out_size, void* d_ws, size_t ws_size,
                              hipStream_t stream) {
  (void)in_sizes; (void)n_in; (void)out_size; (void)ws_size;
  const int Nn = 4, C = 128, H = 96, W = 96;
  const long HW = (long)H * W, CHW = (long)C * HW, PER = Nn * CHW;

#define IN(i) ((const float*)d_in[i])
  // leaf order = jax tree-flatten (sorted dict keys); bn tuple kept in order
  const float* gr_f = IN(0);
  const float* gt_f = IN(1);
  struct Cbr { const float *b,*g,*be,*m,*v,*w; };
  auto cbr = [&](int i) { return Cbr{IN(i), IN(i+1), IN(i+2), IN(i+3), IN(i+4), IN(i+5)}; };
  const float *alpha_b = IN(2), *alpha_w = IN(3);
  const float *bata_b  = IN(4), *bata_w  = IN(5);
  Cbr combine = cbr(6);
  Cbr db1c1 = cbr(12), db1c2 = cbr(18);
  Cbr db2c1 = cbr(24), db2c2 = cbr(30);
  Cbr db3c1 = cbr(36), db3c2 = cbr(42);
  const float *deta_b = IN(48), *deta_w = IN(49);
  const float *dyn_b  = IN(50), *dyn_w  = IN(51);
  Cbr fus1 = cbr(52), fus2 = cbr(58), fus3 = cbr(64);

  // ----- workspace carve -----
  char* wp = (char*)d_ws;
  bf16* gr_b = (bf16*)carve(wp, PER * 2);
  bf16* gt_b = (bf16*)carve(wp, PER * 2);
  bf16* x0   = (bf16*)carve(wp, PER * 4);           // [N,256,96,96]
  bf16* cat1 = (bf16*)carve(wp, PER * 4);           // [gr | warped gt]
  bf16* t1   = (bf16*)carve(wp, (long)Nn*128*48*48*2);
  bf16* n1   = (bf16*)carve(wp, (long)Nn*128*48*48*2);
  bf16* t2   = (bf16*)carve(wp, (long)Nn*128*24*24*2);
  bf16* n2   = (bf16*)carve(wp, (long)Nn*128*24*24*2);
  bf16* t3   = (bf16*)carve(wp, (long)Nn*128*12*12*2);
  bf16* n3   = (bf16*)carve(wp, (long)Nn*128*12*12*2);
  float* pooled = (float*)carve(wp, Nn*128*4);
  float* theta  = (float*)carve(wp, Nn*6*4);
  bf16* modb   = (bf16*)carve(wp, PER * 2);
  bf16* dynout = (bf16*)carve(wp, PER * 2);
  bf16* cat3   = (bf16*)carve(wp, (long)Nn*192*HW*2);

  struct Lay { bf16* w; float* sc; float* bs; int OC; int K; };
  auto mklay = [&](int OC, int K) {
    Lay L; L.w = (bf16*)carve(wp, (long)OC*K*2);
    L.sc = (float*)carve(wp, OC*4); L.bs = (float*)carve(wp, OC*4);
    L.OC = OC; L.K = K; return L;
  };
  Lay Ldb1c1 = mklay(128, 256*9), Ldb1c2 = mklay(128, 128);
  Lay Ldb2c1 = mklay(128, 128*9), Ldb2c2 = mklay(128, 128);
  Lay Ldb3c1 = mklay(128, 128*9), Ldb3c2 = mklay(128, 128);
  Lay Lfus1  = mklay(64, 256), Lfus2 = mklay(64, 128), Lfus3 = mklay(64, 128);
  Lay Lcomb  = mklay(128, 192*9);
  bf16* Wdyn = (bf16*)carve(wp, (long)1152*1152*2);

  auto cvt = [&](const float* s, bf16* d, long n) {
    k_cvt<<<dim3((unsigned)((n + 255) / 256)), dim3(256), 0, stream>>>(s, d, n);
  };
  auto prepCbr = [&](const Cbr& p, const Lay& L) {
    cvt(p.w, L.w, (long)L.OC * L.K);
    k_bncoef<<<dim3((L.OC + 63) / 64), dim3(64), 0, stream>>>(
        p.b, p.g, p.be, p.m, p.v, L.sc, L.bs, L.OC);
  };

  // ----- stage 0: conversions -----
  k_prep<<<dim3((unsigned)((PER + 255) / 256)), dim3(256), 0, stream>>>(
      gr_f, gt_f, gr_b, gt_b, x0, cat1, PER, CHW);
  prepCbr(db1c1, Ldb1c1); prepCbr(db1c2, Ldb1c2);
  prepCbr(db2c1, Ldb2c1); prepCbr(db2c2, Ldb2c2);
  prepCbr(db3c1, Ldb3c1); prepCbr(db3c2, Ldb3c2);
  prepCbr(fus1, Lfus1); prepCbr(fus2, Lfus2); prepCbr(fus3, Lfus3);
  prepCbr(combine, Lcomb);
  cvt(dyn_w, Wdyn, (long)1152 * 1152);

  auto conv3s2 = [&](const bf16* in, bf16* out, const Lay& L, int IC, int IH) {
    int OH = IH / 2;
    dim3 grid((unsigned)(Nn * OH * OH / 16), 1);
    k_conv_wmma<3, 2, 1, true><<<grid, dim3(256), 0, stream>>>(
        in, out, nullptr, L.w, L.sc, L.bs, Nn, IC, L.OC, IH, IH, OH, OH, L.OC, 0);
  };
  auto conv1 = [&](const bf16* in, bf16* out, const Lay& L, int IC, int IH,
                   int outCtot, int ocOff) {
    int nWav = L.OC / 16 > 8 ? 8 : L.OC / 16;
    dim3 grid((unsigned)(Nn * IH * IH / 16), (unsigned)(L.OC / (16 * nWav)));
    k_conv_wmma<1, 1, 0, true><<<grid, dim3(nWav * 32), 0, stream>>>(
        in, out, nullptr, L.w, L.sc, L.bs, Nn, IC, L.OC, IH, IH, IH, IH, outCtot, ocOff);
  };

  // ----- down path -----
  conv3s2(x0, t1, Ldb1c1, 256, 96);  conv1(t1, n1, Ldb1c2, 128, 48, 128, 0);
  conv3s2(n1, t2, Ldb2c1, 128, 48);  conv1(t2, n2, Ldb2c2, 128, 24, 128, 0);
  conv3s2(n2, t3, Ldb3c1, 128, 24);  conv1(t3, n3, Ldb3c2, 128, 12, 128, 0);

  // ----- STN -----
  k_pool<<<dim3(2), dim3(256), 0, stream>>>(n3, pooled, Nn, 128, 12);
  k_theta<<<dim3(1), dim3(32), 0, stream>>>(pooled, deta_w, deta_b, theta, Nn);
  k_warp<<<dim3((unsigned)(Nn * HW / 256)), dim3(256), 0, stream>>>(
      gt_f, theta, cat1, Nn, C, H, W);

  // ----- three fusion branches (write into cat3 channel slices) -----
  conv1(cat1, cat3, Lfus1, 256, 96, 192, 0);                      // in1
  k_mod<<<dim3((unsigned)(Nn * HW / 256)), dim3(256), 0, stream>>>(
      gr_f, gt_f, alpha_w, alpha_b, bata_w, bata_b, modb, Nn, C, H, W);
  conv1(modb, cat3, Lfus2, 128, 96, 192, 64);                     // in2
  {
    size_t lds = (size_t)1152 * 16 * sizeof(float) + 2 * 16 * 34 * sizeof(bf16);
    k_dyn_fused<<<dim3((unsigned)(Nn * HW / 16)), dim3(256), lds, stream>>>(
        gr_b, gt_b, Wdyn, dyn_b, dynout, Nn, H, W);
  }
  conv1(dynout, cat3, Lfus3, 128, 96, 192, 128);                  // in3

  // ----- combine 3x3 -> f32 output -----
  {
    dim3 grid((unsigned)(Nn * HW / 16), 1);
    k_conv_wmma<3, 1, 1, true><<<grid, dim3(256), 0, stream>>>(
        cat3, nullptr, (float*)d_out, Lcomb.w, Lcomb.sc, Lcomb.bs,
        Nn, 192, 128, H, W, H, W, 128, 0);
  }
#undef IN
}
